// VectorQuantizer_67843303408025
// MI455X (gfx1250) — matrix operand order repaired and checked by reading, not compiled, not run on hardware
//
#include <hip/hip_runtime.h>
#include <stdint.h>

// Problem constants (from reference setup_inputs)
#define Bv   16
#define Cv   256
#define HWv  1024          // H*W
#define Nv   16384         // B*H*W query vectors
#define Kv   8192          // codebook size
#define TILESv 512         // Kv / 16 code tiles
#define CHUNKSv 8          // grid split over code dim (parallelism)
#define TILES_PER_CHUNK (TILESv / CHUNKSv)   // 64
#define SCALEv 8192.0f     // codebook pre-scale into f16 sweet spot
#define INV2Sv (2.0f / 8192.0f)
#define LDSB_STRIDE 264    // 256 + 8 halves pad -> conflict-free column reads

// CDNA5 async global->LDS path (verified on this toolchain; signature:
// (AS1 int4*, AS3 int4*, imm offset, imm cpol))
#if defined(__has_builtin)
#if __has_builtin(__builtin_amdgcn_global_load_async_to_lds_b128)
#define VQ_ASYNC 1
#endif
#endif
#ifndef VQ_ASYNC
#define VQ_ASYNC 0
#endif

typedef int vq_v4i __attribute__((vector_size(16)));
#define AS1P(p) ((__attribute__((address_space(1))) vq_v4i*)(p))
#define AS3P(p) ((__attribute__((address_space(3))) vq_v4i*)(p))

__device__ __forceinline__ void vq_wait_asynccnt0() {
#if defined(__has_builtin)
#if __has_builtin(__builtin_amdgcn_s_wait_asynccnt)
  __builtin_amdgcn_s_wait_asynccnt(0);
  return;
#else
  asm volatile("s_wait_asynccnt 0x0" ::: "memory");
  return;
#endif
#else
  asm volatile("s_wait_asynccnt 0x0" ::: "memory");
#endif
}

typedef __attribute__((ext_vector_type(16))) _Float16 v16h;
typedef __attribute__((ext_vector_type(8)))  _Float16 v8h;
typedef __attribute__((ext_vector_type(8)))  float    v8f;

__device__ __forceinline__ v16h cat8h(v8h lo, v8h hi) {
  return __builtin_shufflevector(lo, hi, 0,1,2,3,4,5,6,7,8,9,10,11,12,13,14,15);
}

// ---------------------------------------------------------------------------
// Kernel 0: init packed argmin keys to +inf
// ---------------------------------------------------------------------------
__global__ __launch_bounds__(256) void vq_init_keys(unsigned long long* __restrict__ keys) {
  int i = blockIdx.x * 256 + threadIdx.x;
  if (i < Nv) keys[i] = 0xFFFFFFFFFFFFFFFFull;
}

// ---------------------------------------------------------------------------
// Kernel 1: z [B,C,H,W] f32 -> zf16 [N, C] f16 (LDS-tiled transpose)
// ---------------------------------------------------------------------------
__global__ __launch_bounds__(256) void vq_cvt_z(const float* __restrict__ z,
                                                _Float16* __restrict__ zf16) {
  __shared__ float sh[32][33];
  const int bid = blockIdx.x;
  const int b   = bid >> 8;
  const int rem = bid & 255;
  const int c0  = (rem >> 5) << 5;
  const int hw0 = (rem & 31) << 5;
  const int t = threadIdx.x;
  #pragma unroll
  for (int j = 0; j < 4; ++j) {
    int p = t + 256 * j;
    int cl = p >> 5, hwl = p & 31;
    sh[cl][hwl] = z[(size_t)b * (Cv * HWv) + (size_t)(c0 + cl) * HWv + hw0 + hwl];
  }
  __syncthreads();
  #pragma unroll
  for (int j = 0; j < 4; ++j) {
    int p = t + 256 * j;
    int cl = p & 31, hwl = p >> 5;
    zf16[(size_t)(b * HWv + hw0 + hwl) * Cv + c0 + cl] = (_Float16)sh[cl][hwl];
  }
}

// ---------------------------------------------------------------------------
// Kernel 2: embedding [K,C] f32 -> ef16 (scaled) + enorm[k] = sum(e^2) (fp32)
// ---------------------------------------------------------------------------
__global__ __launch_bounds__(256) void vq_cvt_e(const float* __restrict__ emb,
                                                _Float16* __restrict__ ef16,
                                                float* __restrict__ enorm) {
  const int t = threadIdx.x;
  const int lane = t & 31, wave = t >> 5;
  const int row = blockIdx.x * 8 + wave;
  const float* er = emb + (size_t)row * Cv;
  _Float16* eo = ef16 + (size_t)row * Cv;
  float ss = 0.f;
  #pragma unroll
  for (int j = 0; j < 8; ++j) {
    int c = lane + 32 * j;
    float v = er[c];
    ss += v * v;
    eo[c] = (_Float16)(v * SCALEv);
  }
  #pragma unroll
  for (int off = 16; off >= 1; off >>= 1) ss += __shfl_xor(ss, off, 32);
  if (lane == 0) enorm[row] = ss;
}

// ---------------------------------------------------------------------------
// Kernel 3: fused GEMM + argmin, double-buffered async-LDS pipeline.
// grid = (Nv/128, CHUNKSv); block = 256 (8 wave32s). Wave w owns 16 rows with
// A fragments resident in VGPRs; the block stages 16-code x 256 B-tiles into
// LDS via global_load_async_to_lds_b128, one tile ahead of compute.
// score_k = ||e_k||^2 - (2/S)*dot_scaled; chunks merge via packed atomicMin.
// ---------------------------------------------------------------------------
__global__ __launch_bounds__(256) void vq_argmin(const _Float16* __restrict__ zf16,
                                                 const _Float16* __restrict__ ef16,
                                                 const float* __restrict__ enorm,
                                                 unsigned long long* __restrict__ keys) {
  __shared__ _Float16 ldsB[2][16 * LDSB_STRIDE];
  const int t = threadIdx.x;
  const int lane = t & 31;
  const int wave = t >> 5;
  const int rowBase = blockIdx.x * 128 + wave * 16;
  const int col = lane & 15;             // C/D: N index within tile
  const int hi  = lane >> 4;             // lane-half select

  // Staging map: thread -> (local code row, 16-half segment); 8 KB per tile.
  const int sr   = t >> 4;
  const int sseg = (t & 15) * 16;
  const int tile0 = blockIdx.y * TILES_PER_CHUNK;

  // A fragments: 16 rows x 256 c, resident (8 x v16h = 64 VGPRs).
  // ISA 16-bit A layout: lanes 0-15: M=lane, K={cb+0..7, cb+16..23};
  //                      lanes 16-31: M=lane-16, K={cb+8..15, cb+24..31}.
  v16h afrag[8];
  {
    const _Float16* zrow = zf16 + (size_t)(rowBase + col) * Cv;
    #pragma unroll
    for (int s = 0; s < 8; ++s) {
      const int cb = s * 32;
      v8h lo = *(const v8h*)(zrow + cb + hi * 8);
      v8h hh = *(const v8h*)(zrow + cb + 16 + hi * 8);
      afrag[s] = cat8h(lo, hh);
    }
  }

  float bestVal[8];
  int   bestIdx[8];
  #pragma unroll
  for (int g = 0; g < 8; ++g) { bestVal[g] = 3.402823466e+38f; bestIdx[g] = 0; }

#if VQ_ASYNC
  // ---- async global->LDS pipeline (ASYNCcnt-tracked) ----
  {
    const _Float16* src = ef16 + (size_t)(tile0 * 16 + sr) * Cv + sseg;
    _Float16* dst = &ldsB[0][sr * LDSB_STRIDE + sseg];
    __builtin_amdgcn_global_load_async_to_lds_b128(AS1P(src), AS3P(dst), 0, 0);
    __builtin_amdgcn_global_load_async_to_lds_b128(AS1P(src + 8), AS3P(dst + 8), 0, 0);
  }
  int p = 0;
  for (int it = 0; it < TILES_PER_CHUNK; ++it) {
    const int tile = tile0 + it;
    vq_wait_asynccnt0();     // own async stores landed in LDS
    __syncthreads();         // everyone's stores landed; prev compute done
    if (it + 1 < TILES_PER_CHUNK) {
      const _Float16* src = ef16 + (size_t)((tile + 1) * 16 + sr) * Cv + sseg;
      _Float16* dst = &ldsB[p ^ 1][sr * LDSB_STRIDE + sseg];
      __builtin_amdgcn_global_load_async_to_lds_b128(AS1P(src), AS3P(dst), 0, 0);
      __builtin_amdgcn_global_load_async_to_lds_b128(AS1P(src + 8), AS3P(dst + 8), 0, 0);
      // pull a future tile into WGP$/L2 (locality 3 -> WGP scope)
      int pft = (tile + 4 < TILESv) ? tile + 4 : TILESv - 1;
      __builtin_prefetch(ef16 + (size_t)(pft * 16 + sr) * Cv + sseg, 0, 3);
    }
#else
  // ---- fallback: register-staged double-buffered pipeline ----
  uint4 st0, st1;
  {
    const _Float16* src = ef16 + (size_t)(tile0 * 16 + sr) * Cv + sseg;
    st0 = *(const uint4*)(src);
    st1 = *(const uint4*)(src + 8);
    *(uint4*)(&ldsB[0][sr * LDSB_STRIDE + sseg])     = st0;
    *(uint4*)(&ldsB[0][sr * LDSB_STRIDE + sseg + 8]) = st1;
  }
  __syncthreads();
  int p = 0;
  for (int it = 0; it < TILES_PER_CHUNK; ++it) {
    const int tile = tile0 + it;
    if (it + 1 < TILES_PER_CHUNK) {
      const _Float16* src = ef16 + (size_t)((tile + 1) * 16 + sr) * Cv + sseg;
      st0 = *(const uint4*)(src);        // in flight during compute
      st1 = *(const uint4*)(src + 8);
      int pft = (tile + 4 < TILESv) ? tile + 4 : TILESv - 1;
      __builtin_prefetch(ef16 + (size_t)(pft * 16 + sr) * Cv + sseg, 0, 3);
    }
#endif

    // ---- compute tile from buffer p ----
    // ISA 16-bit B layout (32x16): lanes 0-15: N=lane, K={cb+0..7, cb+16..23};
    //                              lanes 16-31: N=lane-16, K={cb+8..15, cb+24..31}.
    v16h bfrag[8];
    const _Float16* bbase = &ldsB[p][col * LDSB_STRIDE + hi * 8];
    #pragma unroll
    for (int s = 0; s < 8; ++s) {
      v8h lo = *(const v8h*)(bbase + s * 32);
      v8h hh = *(const v8h*)(bbase + s * 32 + 16);
      bfrag[s] = cat8h(lo, hh);
    }
    v8f acc = {};
    #pragma unroll
    for (int s = 0; s < 8; ++s) {
      acc = __builtin_amdgcn_wmma_f32_16x16x32_f16(
          /*neg_a=*/false, afrag[s], /*neg_b=*/false, bfrag[s],
          /*c_mod=*/(short)0, acc, /*reuse_a=*/false, /*reuse_b=*/false);
    }

    const int kc = tile * 16 + col;
    const float en = enorm[kc];
    // C/D layout: vgpr g -> row (g + 8*hi), col = lane&15
    #pragma unroll
    for (int g = 0; g < 8; ++g) {
      float score = en - INV2Sv * acc[g];
      if (score < bestVal[g]) { bestVal[g] = score; bestIdx[g] = kc; }
    }

#if VQ_ASYNC
    p ^= 1;
  }
#else
    if (it + 1 < TILES_PER_CHUNK) {
      *(uint4*)(&ldsB[p ^ 1][sr * LDSB_STRIDE + sseg])     = st0;
      *(uint4*)(&ldsB[p ^ 1][sr * LDSB_STRIDE + sseg + 8]) = st1;
    }
    __syncthreads();
    p ^= 1;
  }
#endif

  // Reduce over the 16 lane-columns holding the same rows.
  #pragma unroll
  for (int off = 1; off < 16; off <<= 1) {
    #pragma unroll
    for (int g = 0; g < 8; ++g) {
      float ov = __shfl_xor(bestVal[g], off, 32);
      int   oi = __shfl_xor(bestIdx[g], off, 32);
      if (ov < bestVal[g] || (ov == bestVal[g] && oi < bestIdx[g])) {
        bestVal[g] = ov; bestIdx[g] = oi;
      }
    }
  }
  if (col == 0) {
    const int rbase = rowBase + hi * 8;
    #pragma unroll
    for (int g = 0; g < 8; ++g) {
      unsigned int fb = __float_as_uint(bestVal[g]);
      fb = (fb & 0x80000000u) ? ~fb : (fb | 0x80000000u);   // order-preserving
      unsigned long long key =
          ((unsigned long long)fb << 32) | (unsigned int)bestIdx[g];
      atomicMin(&keys[rbase + g], key);   // ties -> smaller idx wins
    }
  }
}

// ---------------------------------------------------------------------------
// Kernel 4: decode packed keys -> int idx (ws) + float idx (d_out tail)
// ---------------------------------------------------------------------------
__global__ __launch_bounds__(256) void vq_finalize(const unsigned long long* __restrict__ keys,
                                                   int* __restrict__ idxi,
                                                   float* __restrict__ idxf) {
  int i = blockIdx.x * 256 + threadIdx.x;
  if (i < Nv) {
    int id = (int)(keys[i] & 0xFFFFFFFFull);
    idxi[i] = id;
    idxf[i] = (float)id;
  }
}

// ---------------------------------------------------------------------------
// Kernel 5: epilogue. z_q [B,C,H,W] = embedding[idx] (fp32 exact);
// loss [B,H,W,C] = 1.25*(z_q - z)^2. LDS-tiled, all streams coalesced.
// ---------------------------------------------------------------------------
__global__ __launch_bounds__(256) void vq_epilogue(const float* __restrict__ z,
                                                   const float* __restrict__ emb,
                                                   const int* __restrict__ idx,
                                                   float* __restrict__ zq,
                                                   float* __restrict__ loss) {
  __shared__ float sh_e[32][33];
  __shared__ float sh_l[32][33];
  __shared__ int   sh_i[32];
  const int bid = blockIdx.x;
  const int b   = bid >> 8;
  const int rem = bid & 255;
  const int c0  = (rem >> 5) << 5;
  const int hw0 = (rem & 31) << 5;
  const int t = threadIdx.x;

  if (t < 32) sh_i[t] = idx[b * HWv + hw0 + t];
  __syncthreads();

  #pragma unroll
  for (int j = 0; j < 4; ++j) {
    int p = t + 256 * j;
    int hwl = p >> 5, cl = p & 31;
    sh_e[hwl][cl] = emb[(size_t)sh_i[hwl] * Cv + c0 + cl];
  }
  __syncthreads();

  #pragma unroll
  for (int j = 0; j < 4; ++j) {
    int p = t + 256 * j;
    int cl = p >> 5, hwl = p & 31;
    size_t zi = (size_t)b * (Cv * HWv) + (size_t)(c0 + cl) * HWv + hw0 + hwl;
    float zv = z[zi];
    float ev = sh_e[hwl][cl];
    zq[zi] = ev;
    float d = ev - zv;
    sh_l[hwl][cl] = 1.25f * d * d;
  }
  __syncthreads();

  #pragma unroll
  for (int j = 0; j < 4; ++j) {
    int p = t + 256 * j;
    int hwl = p >> 5, cl = p & 31;
    loss[(size_t)(b * HWv + hw0 + hwl) * Cv + c0 + cl] = sh_l[hwl][cl];
  }
}

// ---------------------------------------------------------------------------
extern "C" void kernel_launch(void* const* d_in, const int* in_sizes, int n_in,
                              void* d_out, int out_size, void* d_ws, size_t ws_size,
                              hipStream_t stream) {
  const float* z   = (const float*)d_in[0];   // [16,256,32,32]
  const float* emb = (const float*)d_in[1];   // [8192,256]

  // Workspace layout (~12.8 MB total; assumes ws_size is at least this).
  char* ws = (char*)d_ws;
  const size_t OFF_ZF16  = 0;                                   // 8 MB
  const size_t OFF_EF16  = OFF_ZF16 + (size_t)Nv * Cv * 2;      // +4 MB
  const size_t OFF_ENORM = OFF_EF16 + (size_t)Kv * Cv * 2;      // +32 KB
  const size_t OFF_IDX   = OFF_ENORM + (size_t)Kv * 4;          // +64 KB
  const size_t OFF_KEYS  = OFF_IDX + (size_t)Nv * 4;            // +128 KB
  _Float16* zf16  = (_Float16*)(ws + OFF_ZF16);
  _Float16* ef16  = (_Float16*)(ws + OFF_EF16);
  float*    enorm = (float*)(ws + OFF_ENORM);
  int*      idxi  = (int*)(ws + OFF_IDX);
  unsigned long long* keys = (unsigned long long*)(ws + OFF_KEYS);

  // d_out: [z_q 4194304 f32][loss 4194304 f32][idx 16384 as f32]
  float* out_zq   = (float*)d_out;
  float* out_loss = out_zq + (size_t)Bv * Cv * HWv;
  float* out_idx  = out_loss + (size_t)Bv * Cv * HWv;

  vq_init_keys<<<Nv / 256, 256, 0, stream>>>(keys);
  vq_cvt_z<<<Bv * (Cv / 32) * (HWv / 32), 256, 0, stream>>>(z, zf16);
  vq_cvt_e<<<Kv / 8, 256, 0, stream>>>(emb, ef16, enorm);
  vq_argmin<<<dim3(Nv / 128, CHUNKSv), 256, 0, stream>>>(zf16, ef16, enorm, keys);
  vq_finalize<<<Nv / 256, 256, 0, stream>>>(keys, idxi, out_idx);
  vq_epilogue<<<Bv * (Cv / 32) * (HWv / 32), 256, 0, stream>>>(z, emb, idxi, out_zq, out_loss);
}
